// CSMoE_40389872451637
// MI455X (gfx1250) — matrix operand order repaired and checked by reading, not compiled, not run on hardware
//
#include <hip/hip_runtime.h>
#include <hip/hip_bf16.h>

// CDNA5 (gfx1250) routed top-2 MoE FFN with bf16 WMMA (fp32 accumulate).
//
// Inputs (setup_inputs order):
//  0: x   [B,S,D]   f32      4: W1 [E,D,FFN] f32
//  1: ctx [B,S,D]   f32      5: b1 [E,FFN]   f32
//  2: Wg  [D,E]     f32      6: W2 [E,FFN,D] f32
//  3: bg  [E]       f32      7: b2 [E,D]     f32
// Output: [B,S,D] f32
//
// Roofline: routed (top-2) matrix work = 69 GFLOP; mandatory weight traffic
// = 256 MB fp32 read once (~11 us @ 23.3 TB/s). bf16 WMMA w/ f32 accumulate
// keeps compute (~15 us class) balanced against HBM; fp32 weights are
// converted to bf16 in-register (no extra global traffic).

typedef __bf16 bf16;
typedef __attribute__((ext_vector_type(16))) __bf16 v16bf;
typedef __attribute__((ext_vector_type(8)))  float  v8f;

#define NDM   1024   // D_MODEL
#define NFFN  4096
#define NE    8
#define NTOK  2048   // B*S
#define MT    16     // tokens per tile
#define FCH   128    // FFN chunk per outer iteration
#define MAXT  (NTOK / MT)   // 128 worst-case tiles per expert

// ---------------- kernel 0: zero output + expert counters ------------------
__global__ void moe_zero(float* __restrict__ out, int n, int* __restrict__ counts) {
    int i = blockIdx.x * blockDim.x + threadIdx.x;
    if (i < n) out[i] = 0.0f;
    if (i < NE) counts[i] = 0;
}

// ---------------- kernel 1: gating (logits -> top2 -> softmax -> buckets) ---
__global__ __launch_bounds__(256) void moe_gate(
    const float* __restrict__ ctx, const float* __restrict__ Wg,
    const float* __restrict__ bg,  int* __restrict__ counts,
    int* __restrict__ tlist, float* __restrict__ tw) {
    int t = blockIdx.x * blockDim.x + threadIdx.x;
    if (t >= NTOK) return;
    const float* c = ctx + (size_t)t * NDM;
    float lg[NE];
#pragma unroll
    for (int e = 0; e < NE; ++e) lg[e] = bg[e];
    for (int d = 0; d < NDM; ++d) {
        float cv = c[d];
        const float* w = Wg + (size_t)d * NE;
#pragma unroll
        for (int e = 0; e < NE; ++e) lg[e] += cv * w[e];
    }
    int i0 = 0; float v0 = lg[0];
#pragma unroll
    for (int e = 1; e < NE; ++e) if (lg[e] > v0) { v0 = lg[e]; i0 = e; }
    int i1 = -1; float v1 = -3.4e38f;
#pragma unroll
    for (int e = 0; e < NE; ++e) if (e != i0 && lg[e] > v1) { v1 = lg[e]; i1 = e; }
    float w1e = __expf(v1 - v0);        // softmax over {v0, v1}, max-shifted
    float inv = 1.0f / (1.0f + w1e);
    float w0 = inv, w1 = w1e * inv;
    int s0 = atomicAdd(&counts[i0], 1);
    tlist[i0 * NTOK + s0] = t; tw[i0 * NTOK + s0] = w0;
    int s1 = atomicAdd(&counts[i1], 1);
    tlist[i1 * NTOK + s1] = t; tw[i1 * NTOK + s1] = w1;
}

// ---------------- kernel 2: grouped expert FFN with bf16 WMMA ---------------
// Block = 256 threads = 8 wave32. One block handles one 16-token tile of one
// expert. Wave w owns FFN columns [fc+16w, fc+16w+16) in phase 1 and D columns
// [128w, 128w+128) (8 WMMA N-subtiles) in phase 2. Grid walks tiles of the
// same expert consecutively so the 32 MB expert weight slice stays L2-resident.
__global__ __launch_bounds__(256) void moe_ffn(
    const float* __restrict__ x,  const float* __restrict__ W1,
    const float* __restrict__ b1, const float* __restrict__ W2,
    const float* __restrict__ b2, const int* __restrict__ counts,
    const int* __restrict__ tlist, const float* __restrict__ tw,
    float* __restrict__ out) {

    __shared__ bf16  sX[MT * NDM];   // gathered x tile, bf16 (32 KB)
    __shared__ bf16  sH[MT * FCH];   // relu(h) chunk, bf16   (4 KB)
    __shared__ int   sTok[MT];
    __shared__ float sWt[MT];

    const int e    = blockIdx.y;
    const int tile = blockIdx.x;
    const int cnt  = counts[e];
    if (tile * MT >= cnt) return;              // block-uniform: EXEC stays full

    const int tid  = threadIdx.x;
    const int lane = tid & 31;
    const int wave = tid >> 5;

    if (tid < MT) {
        int idx = tile * MT + tid;
        if (idx < cnt) { sTok[tid] = tlist[e * NTOK + idx]; sWt[tid] = tw[e * NTOK + idx]; }
        else           { sTok[tid] = -1;                    sWt[tid] = 0.0f; }
    }
    __syncthreads();

    // Gather + fp32->bf16 convert the 16 x rows into LDS (zeros for padding).
    // Rows are contiguous and 4 KB aligned: use float4 loads (b128).
    for (int i = tid * 4; i < MT * NDM; i += 256 * 4) {
        int m = i >> 10, d = i & (NDM - 1);
        int t = sTok[m];
        float4 v = make_float4(0.f, 0.f, 0.f, 0.f);
        if (t >= 0) v = *(const float4*)(x + (size_t)t * NDM + d);
        bf16* dst = &sX[i];
        dst[0] = (bf16)v.x; dst[1] = (bf16)v.y;
        dst[2] = (bf16)v.z; dst[3] = (bf16)v.w;
    }
    __syncthreads();

    const float* W1e = W1 + (size_t)e * NDM * NFFN;
    const float* W2e = W2 + (size_t)e * NFFN * NDM;

    // Lane striping this code uses for 16-bit operands (wave32, contiguous k in A and B):
    //  A 16x32: lanes 0-15 row M=lane hold K = 0..15; lanes 16-31 hold
    //           row M=lane-16, K = 16..31.
    //  B 32x16: lanes 0-15 col N=lane hold K = 0..15; lanes 16-31 K = 16..31.
    //  C/D 16x16 f32: lanes 0-15 vgpr j -> (M=j, N=lane); lanes 16-31 -> M=8+j.
    const int r   = lane & 15;
    const int hi  = lane >> 4;
    const int kbA = hi * 8;
    const int kbB = hi * 16;

    v8f acc[8];
#pragma unroll
    for (int st = 0; st < 8; ++st) acc[st] = (v8f){};

    for (int fc = 0; fc < NFFN; fc += FCH) {
        // ---- phase 1: h[16,16] = relu(xTile @ W1[:, fcol..fcol+16) + b1) ----
        v8f hacc = (v8f){};
        const int fcol = fc + wave * 16 + r;
        // unroll x2: two independent load/convert streams overlap the XDL pipe
#pragma unroll 2
        for (int kk = 0; kk < NDM; kk += 32) {
            v16bf a, b;
#pragma unroll
            for (int j = 0; j < 8; ++j) {
                a[j]     = sX[r * NDM + kk + kbB + j];
                a[j + 8] = sX[r * NDM + kk + kbB + 8 + j];
            }
            const float* wp = W1e + (size_t)(kk + kbB) * NFFN + fcol;
            __builtin_prefetch(wp + (size_t)32 * NFFN, 0, 0);   // global_prefetch_b8
#pragma unroll
            for (int j = 0; j < 16; ++j) b[j] = (bf16)wp[(size_t)j * NFFN];
            hacc = __builtin_amdgcn_wmma_f32_16x16x32_bf16(
                false, a, false, b, (short)0, hacc, false, false);
        }
        {   // +b1, ReLU, bf16 -> LDS (row-major 16 x 128)
            const int n = wave * 16 + r;
            const float bias = b1[e * NFFN + fc + n];
#pragma unroll
            for (int j = 0; j < 8; ++j) {
                float v = hacc[j] + bias;
                v = v > 0.0f ? v : 0.0f;
                sH[(hi * 8 + j) * FCH + n] = (bf16)v;
            }
        }
        __syncthreads();

        // ---- phase 2: acc[16,128] += h[16,128] @ W2[fc..fc+128, dcols) ----
#pragma unroll 1
        for (int kk = 0; kk < FCH; kk += 32) {
            v16bf a;
#pragma unroll
            for (int j = 0; j < 8; ++j) {
                a[j]     = sH[r * FCH + kk + kbB + j];
                a[j + 8] = sH[r * FCH + kk + kbB + 8 + j];
            }
            const float* w2p = W2e + (size_t)(fc + kk + kbB) * NDM + wave * 128 + r;
#pragma unroll
            for (int st = 0; st < 8; ++st) {
                v16bf b;
#pragma unroll
                for (int j = 0; j < 16; ++j) b[j] = (bf16)w2p[(size_t)j * NDM + st * 16];
                acc[st] = __builtin_amdgcn_wmma_f32_16x16x32_bf16(
                    false, a, false, b, (short)0, acc[st], false, false);
            }
        }
        __syncthreads();   // before next chunk overwrites sH
    }

    // ---- epilogue: +b2, scale by combine weight, atomic add into out ----
#pragma unroll
    for (int st = 0; st < 8; ++st) {
        const int col = wave * 128 + st * 16 + r;
        const float bias = b2[e * NDM + col];
#pragma unroll
        for (int j = 0; j < 8; ++j) {
            int row = hi * 8 + j;
            int t = sTok[row];
            if (t >= 0)
                atomicAdd(&out[(size_t)t * NDM + col], sWt[row] * (acc[st][j] + bias));
        }
    }
}

// ---------------------------- launcher --------------------------------------
extern "C" void kernel_launch(void* const* d_in, const int* in_sizes, int n_in,
                              void* d_out, int out_size, void* d_ws, size_t ws_size,
                              hipStream_t stream) {
    const float* x   = (const float*)d_in[0];
    const float* ctx = (const float*)d_in[1];
    const float* Wg  = (const float*)d_in[2];
    const float* bg  = (const float*)d_in[3];
    const float* W1  = (const float*)d_in[4];
    const float* b1  = (const float*)d_in[5];
    const float* W2  = (const float*)d_in[6];
    const float* b2  = (const float*)d_in[7];
    float* out = (float*)d_out;

    // workspace: [counts: 64 ints][tlist: E*NTOK ints][tw: E*NTOK floats] ~131KB
    int*   counts = (int*)d_ws;
    int*   tlist  = counts + 64;
    float* tw     = (float*)(tlist + NE * NTOK);

    moe_zero<<<(out_size + 255) / 256, 256, 0, stream>>>(out, out_size, counts);
    moe_gate<<<NTOK / 256, 256, 0, stream>>>(ctx, Wg, bg, counts, tlist, tw);
    moe_ffn<<<dim3(MAXT, NE), 256, 0, stream>>>(x, W1, b1, W2, b2,
                                                counts, tlist, tw, out);
}
